// SSDNmsBoxHead_81509889343684
// MI455X (gfx1250) — compile-verified, matrix operand order checked by reading, not run on hardware
//
#include <hip/hip_runtime.h>
#include <hip/hip_bf16.h>

typedef unsigned int u32;
typedef _Float16 v16h __attribute__((ext_vector_type(16)));
typedef float    v8f  __attribute__((ext_vector_type(8)));
typedef unsigned int u32x4 __attribute__((ext_vector_type(4)));
typedef int          i32x8 __attribute__((ext_vector_type(8)));
typedef int          i32x4 __attribute__((ext_vector_type(4)));

#define NIMG   16
#define NPRIOR 32768
#define NCLS   81
#define NFG    80
#define NCAND  400
#define NOUT   100
#define NBIN   1024
#define TILE_ROWS 64

// ---------------------------------------------------------------------------
// TDM: stage a [64 x 81] f32 tile (contiguous rows, stride 81) into LDS.
// D# group0: count=1, lds_addr, 57-bit global addr, type=2.
// D# group1: data_size=2 (4B), tensor_dim0=81, tensor_dim1=32768,
//            tile_dim0=81, tile_dim1=64, tensor_dim0_stride=81.
// amdgpu-toolchain builtin: 6 args (g0 u32x4, g1 i32x8, i32x4, i32x4, i32x8, cpol)
// ---------------------------------------------------------------------------
__device__ __forceinline__ void tdm_load_tile(const float* gptr, u32 lds_addr) {
  unsigned long long ga = (unsigned long long)gptr;
  u32x4 g0;
  g0[0] = 1u;                                               // count = 1
  g0[1] = lds_addr;                                         // LDS byte address
  g0[2] = (u32)ga;                                          // global addr lo
  g0[3] = ((u32)(ga >> 32) & 0x01FFFFFFu) | (2u << 30);     // addr hi | type=2
  const u32 td0 = NCLS, td1 = NPRIOR, tl0 = NCLS, tl1 = TILE_ROWS;
  const unsigned long long s0 = NCLS;
  i32x8 g1;
  g1[0] = (int)(2u << 16);                                  // data_size = 4B
  g1[1] = (int)((td0 & 0xFFFFu) << 16);                     // tensor_dim0 lo16
  g1[2] = (int)((td0 >> 16) | ((td1 & 0xFFFFu) << 16));     // dim0 hi | dim1 lo
  g1[3] = (int)((td1 >> 16) | (tl0 << 16));                 // dim1 hi | tile_dim0
  g1[4] = (int)(tl1);                                       // tile_dim1, tile_dim2=0
  g1[5] = (int)(u32)s0;                                     // stride0 lo32
  g1[6] = (int)((u32)(s0 >> 32) & 0xFFFFu);                 // stride0 hi | stride1 lo
  g1[7] = 0;
  i32x4 z4 = {0, 0, 0, 0};
  i32x8 z8 = {0, 0, 0, 0, 0, 0, 0, 0};
  __builtin_amdgcn_tensor_load_to_lds(g0, g1, z4, z4, z8, 0);
  __builtin_amdgcn_s_wait_tensorcnt(0);
}

// ---------------------------------------------------------------------------
// Per-prior logsumexp using WMMA row-reduction (A[16x32 f16] x ones = row sums).
// Lanes (L, L+16) cooperate on prior m = L&15; hi = L>>4 selects the K-half
// per the documented 16-bit A-matrix layout.
// ---------------------------------------------------------------------------
__device__ __forceinline__ float tile_lse(const float* __restrict__ row, int hi, int m) {
  float mx = -INFINITY;
  for (int c = hi; c < NCLS; c += 2) mx = fmaxf(mx, row[c]);
  mx = fmaxf(mx, __shfl_xor(mx, 16, 32));

  v16h ones;
#pragma unroll
  for (int j = 0; j < 16; ++j) ones[j] = (_Float16)1.0f;
  v8f acc = {0.f, 0.f, 0.f, 0.f, 0.f, 0.f, 0.f, 0.f};
#pragma unroll
  for (int kb = 0; kb < 96; kb += 32) {            // 3 WMMAs cover K=0..95 (pad>80 with 0)
    v16h a;
#pragma unroll
    for (int j = 0; j < 8; ++j) {
      int base = (j < 4 ? 2 * j : 16 + 2 * (j - 4)) + hi * 8 + kb;
      float e0 = (base     < NCLS) ? __expf(row[base]     - mx) : 0.f;
      float e1 = (base + 1 < NCLS) ? __expf(row[base + 1] - mx) : 0.f;
      a[2 * j]     = (_Float16)e0;
      a[2 * j + 1] = (_Float16)e1;
    }
    acc = __builtin_amdgcn_wmma_f32_16x16x32_f16(false, a, false, ones,
                                                 (short)0, acc, false, false);
  }
  // C/D layout: lane<16 VGPR r holds M=r; lane>=16 VGPR r holds M=r+8.
  int own = m & 7;
  float v = acc[0];
#pragma unroll
  for (int k = 1; k < 8; ++k) v = (own == k) ? acc[k] : v;
  int src = (m < 8) ? m : (m + 8);
  float sumExp = __shfl(v, src, 32);
  return mx + __logf(sumExp);
}

__device__ __forceinline__ u32 mono_key(float s) {
  u32 u = __float_as_uint(s);
  return (u & 0x80000000u) ? ~u : (u | 0x80000000u);
}

// ---------------------------------------------------------------------------
__global__ void init_kernel(u32* ghist, int* cntHi, int* cntEq) {
  int i = blockIdx.x * 256 + threadIdx.x;
  if (i < NIMG * NBIN) ghist[i] = 0u;
  if (i < NIMG) { cntHi[i] = 0; cntEq[i] = 0; }
}

// Pass 1: fused log_softmax + 1024-bin key histogram (per image).
__global__ __launch_bounds__(128) void score_hist_kernel(const float* __restrict__ logits,
                                                         u32* __restrict__ ghist) {
  __shared__ float tile[TILE_ROWS * NCLS];
  __shared__ u32 lh[NBIN];
  const int img = blockIdx.y;
  const int prior0 = blockIdx.x * TILE_ROWS;
  for (int i = threadIdx.x; i < NBIN; i += 128) lh[i] = 0u;
  if (threadIdx.x < 32)
    tdm_load_tile(logits + ((size_t)img * NPRIOR + prior0) * NCLS,
                  (u32)(unsigned long long)&tile[0]);
  __syncthreads();

  const int lane = threadIdx.x & 31, wv = threadIdx.x >> 5;
  const int m = lane & 15, hi = lane >> 4;
  const float* row = tile + (wv * 16 + m) * NCLS;
  const float lse = tile_lse(row, hi, m);
  for (int c = 1 + hi; c <= NFG; c += 2)
    atomicAdd(&lh[mono_key(row[c] - lse) >> 22], 1u);
  __syncthreads();
  for (int i = threadIdx.x; i < NBIN; i += 128)
    if (lh[i]) atomicAdd(&ghist[img * NBIN + i], lh[i]);
}

// Find threshold bin so that (#above) + (need from bin) == 400.
__global__ void thresh_kernel(const u32* ghist, int* thrb, int* above, int* needA) {
  int img = threadIdx.x;
  if (img >= NIMG) return;
  const u32* h = ghist + img * NBIN;
  int cum = 0, b;
  for (b = NBIN - 1; b >= 0; --b) {
    int nb = cum + (int)h[b];
    if (nb >= NCAND) break;
    cum = nb;
  }
  if (b < 0) b = 0;
  thrb[img] = b; above[img] = cum; needA[img] = NCAND - cum;
}

// Pass 2: recompute scores, compact exactly 400 (score, flat_idx) per image.
__global__ __launch_bounds__(128) void compact_kernel(const float* __restrict__ logits,
                                                      const int* thrb, const int* above,
                                                      const int* needA, int* cntHi,
                                                      int* cntEq, float* cs, int* ci) {
  __shared__ float tile[TILE_ROWS * NCLS];
  const int img = blockIdx.y;
  const int prior0 = blockIdx.x * TILE_ROWS;
  if (threadIdx.x < 32)
    tdm_load_tile(logits + ((size_t)img * NPRIOR + prior0) * NCLS,
                  (u32)(unsigned long long)&tile[0]);
  __syncthreads();

  const int lane = threadIdx.x & 31, wv = threadIdx.x >> 5;
  const int m = lane & 15, hi = lane >> 4;
  const int p = prior0 + wv * 16 + m;
  const float* row = tile + (wv * 16 + m) * NCLS;
  const float lse = tile_lse(row, hi, m);
  const int tb = thrb[img], ab = above[img], nd = needA[img];
  for (int c = 1 + hi; c <= NFG; c += 2) {
    float s = row[c] - lse;
    int bin = (int)(mono_key(s) >> 22);
    if (bin > tb) {
      int pos = atomicAdd(&cntHi[img], 1);
      if (pos < NCAND) { cs[img * NCAND + pos] = s; ci[img * NCAND + pos] = p * NFG + (c - 1); }
    } else if (bin == tb) {
      int q = atomicAdd(&cntEq[img], 1);
      if (q < nd) {
        int pos = ab + q;
        cs[img * NCAND + pos] = s; ci[img * NCAND + pos] = p * NFG + (c - 1);
      }
    }
  }
}

// Sort 400 candidates, decode boxes, class-offset greedy NMS, emit top-100.
__global__ __launch_bounds__(512) void nms_kernel(const float* __restrict__ bbox,
                                                  const float* __restrict__ priors,
                                                  const float* __restrict__ cs,
                                                  const int* __restrict__ ci,
                                                  float* __restrict__ out) {
  __shared__ float skey[512];
  __shared__ int   sidx[512];
  __shared__ float BX0[512], BY0[512], BX1[512], BY1[512];
  __shared__ float off[512], red[512];
  __shared__ int   keep[512], scn[512], scn2[512];
  const int img = blockIdx.x, t = threadIdx.x;

  skey[t] = (t < NCAND) ? cs[img * NCAND + t] : -INFINITY;
  sidx[t] = (t < NCAND) ? ci[img * NCAND + t] : 0;
  __syncthreads();

  // bitonic sort, descending by score
  for (int k = 2; k <= 512; k <<= 1)
    for (int j = k >> 1; j > 0; j >>= 1) {
      int ixj = t ^ j;
      if (ixj > t) {
        float a = skey[t], b = skey[ixj];
        bool desc = ((t & k) == 0);
        if (desc ? (a < b) : (a > b)) {
          skey[t] = b; skey[ixj] = a;
          int ia = sidx[t]; sidx[t] = sidx[ixj]; sidx[ixj] = ia;
        }
      }
      __syncthreads();
    }

  // decode boxes
  float x0 = 0.f, y0 = 0.f, x1 = 0.f, y1 = 0.f;
  int label = 0;
  if (t < NCAND) {
    int fi = sidx[t];
    int p = fi / NFG;
    label = fi % NFG + 1;
    const float* L = bbox + ((size_t)img * NPRIOR + p) * 4;
    const float* Pr = priors + (size_t)p * 4;
    float cx = L[0] * 0.1f * Pr[2] + Pr[0];
    float cy = L[1] * 0.1f * Pr[3] + Pr[1];
    float w  = __expf(L[2] * 0.2f) * Pr[2];
    float h  = __expf(L[3] * 0.2f) * Pr[3];
    x0 = (cx - 0.5f * w) * 512.f; y0 = (cy - 0.5f * h) * 512.f;
    x1 = (cx + 0.5f * w) * 512.f; y1 = (cy + 0.5f * h) * 512.f;
  }
  BX0[t] = x0; BY0[t] = y0; BX1[t] = x1; BY1[t] = y1;
  red[t] = (t < NCAND) ? fmaxf(fmaxf(x0, y0), fmaxf(x1, y1)) : -INFINITY;
  __syncthreads();
  for (int s = 256; s > 0; s >>= 1) {
    if (t < s) red[t] = fmaxf(red[t], red[t + s]);
    __syncthreads();
  }
  off[t]  = (t < NCAND) ? (float)label * (red[0] + 1.0f) : 0.f;
  keep[t] = (t < NCAND) ? 1 : 0;
  __syncthreads();

  // greedy NMS (IoU on class-offset boxes)
  for (int i = 0; i < NCAND; ++i) {
    if (keep[i] && t > i && t < NCAND && keep[t]) {
      float ax0 = BX0[i] + off[i], ay0 = BY0[i] + off[i];
      float ax1 = BX1[i] + off[i], ay1 = BY1[i] + off[i];
      float bx0 = BX0[t] + off[t], by0 = BY0[t] + off[t];
      float bx1 = BX1[t] + off[t], by1 = BY1[t] + off[t];
      float iw = fmaxf(fminf(ax1, bx1) - fmaxf(ax0, bx0), 0.f);
      float ih = fmaxf(fminf(ay1, by1) - fmaxf(ay0, by0), 0.f);
      float inter = iw * ih;
      float uni = (ax1 - ax0) * (ay1 - ay0) + (bx1 - bx0) * (by1 - by0) - inter;
      if (inter / fmaxf(uni, 1e-12f) > 0.45f) keep[t] = 0;
    }
    __syncthreads();
  }

  // inclusive scans: kept ranks and not-kept ranks (for -inf padding order)
  scn[t]  = keep[t];
  scn2[t] = (t < NCAND) ? (1 - keep[t]) : 0;
  __syncthreads();
  for (int s = 1; s < 512; s <<= 1) {
    int a = (t >= s) ? scn[t - s] : 0;
    int b = (t >= s) ? scn2[t - s] : 0;
    __syncthreads();
    scn[t] += a; scn2[t] += b;
    __syncthreads();
  }
  const int keptTotal = scn[511];

  if (t < NCAND) {
    int r; float sc;
    if (keep[t]) { r = scn[t] - 1; sc = skey[t]; }
    else         { r = keptTotal + (scn2[t] - 1); sc = -INFINITY; }
    if (r < NOUT) {
      float* ob = out + ((size_t)img * NOUT + r) * 4;
      ob[0] = BX0[t]; ob[1] = BY0[t]; ob[2] = BX1[t]; ob[3] = BY1[t];
      out[NIMG * NOUT * 4 + img * NOUT + r] = (float)label;
      out[NIMG * NOUT * 5 + img * NOUT + r] = sc;
    }
  }
}

extern "C" void kernel_launch(void* const* d_in, const int* in_sizes, int n_in,
                              void* d_out, int out_size, void* d_ws, size_t ws_size,
                              hipStream_t stream) {
  const float* logits = (const float*)d_in[0];
  const float* bbox   = (const float*)d_in[1];
  const float* priors = (const float*)d_in[2];
  float* out = (float*)d_out;

  u32*  ghist = (u32*)d_ws;                  // 16*1024
  int*  thrb  = (int*)(ghist + NIMG * NBIN); // 16
  int*  above = thrb + NIMG;
  int*  needA = above + NIMG;
  int*  cntHi = needA + NIMG;
  int*  cntEq = cntHi + NIMG;
  float* cs   = (float*)(cntEq + NIMG);      // 16*400
  int*   ci   = (int*)(cs + NIMG * NCAND);   // 16*400

  init_kernel<<<(NIMG * NBIN + 255) / 256, 256, 0, stream>>>(ghist, cntHi, cntEq);
  dim3 grid(NPRIOR / TILE_ROWS, NIMG);
  score_hist_kernel<<<grid, 128, 0, stream>>>(logits, ghist);
  thresh_kernel<<<1, 32, 0, stream>>>(ghist, thrb, above, needA);
  compact_kernel<<<grid, 128, 0, stream>>>(logits, thrb, above, needA,
                                           cntHi, cntEq, cs, ci);
  nms_kernel<<<NIMG, 512, 0, stream>>>(bbox, priors, cs, ci, out);
}
